// LSTM_55722905699279
// MI455X (gfx1250) — matrix operand order repaired and checked by reading, not compiled, not run on hardware
//
#include <hip/hip_runtime.h>
#include <hip/hip_bf16.h>

typedef __attribute__((ext_vector_type(16))) _Float16 v16h;
typedef __attribute__((ext_vector_type(8)))  _Float16 v8h;
typedef __attribute__((ext_vector_type(8)))  float    v8f;
typedef __attribute__((ext_vector_type(4)))  int      v4i;

#define T_STEPS 512
#define BATCH   64
#define HID     512
#define NGATE   2048      // 4*H
#define IN0     168
#define KX0     192       // IN0 padded to mult of 32
#define KTOT0   704       // KX0 + HID
#define KX1     1024      // 2*H (layer1 input)
#define KTOT1   1536      // KX1 + HID
#define PRE     168
#define NPAD    176       // PRE padded to mult of 16

// ---------------- workspace layout (bytes) ----------------
#define OFF_XH     ((size_t)0)                         // [T][B][192] f16      = 12,582,912
#define OFF_W0     ((size_t)12582912)                  // [2][2048][704] f16   =  5,767,168
#define OFF_W1     ((size_t)18350080)                  // [2][2048][1536] f16  = 12,582,912
#define OFF_WOUT   ((size_t)30932992)                  // [176][1024] f16      =    360,448
#define OFF_HBUF0  ((size_t)31293440)                  // [2][64][512] f16     =    131,072
#define OFF_HBUF1  ((size_t)31424512)                  // [2][64][512] f16     =    131,072
#define OFF_CTR    ((size_t)31555584)                  // counters             =        256
#define OFF_H1     ((size_t)31555840)                  // [T][B][1024] f16     = 67,108,864
#define OFF_H2     ((size_t)98664704)                  // [T][B][1024] f16     = 67,108,864

#if __has_builtin(__builtin_amdgcn_global_load_async_to_lds_b128) && \
    __has_builtin(__builtin_amdgcn_s_wait_asynccnt)
#define HAVE_ASYNC_LDS 1
typedef v4i __attribute__((address_space(1)))* gas_ptr;   // global v4i32*
typedef v4i __attribute__((address_space(3)))* las_ptr;   // LDS    v4i32*
#else
#define HAVE_ASYNC_LDS 0
#endif

static __device__ __forceinline__ v16h join8(v8h a, v8h b) {
  return __builtin_shufflevector(a, b, 0,1,2,3,4,5,6,7,8,9,10,11,12,13,14,15);
}
static __device__ __forceinline__ float sigm(float x) {
  return 1.0f / (1.0f + __expf(-x));
}
static __device__ __forceinline__ float tanh_fast(float x) {
  x = fminf(15.0f, fmaxf(-15.0f, x));
  float e = __expf(2.0f * x);
  return (e - 1.0f) / (e + 1.0f);
}

// ---------------- packing / init kernels ----------------
__global__ void lstm_zero_kernel(unsigned* __restrict__ p, unsigned nwords) {
  unsigned i = blockIdx.x * 256u + threadIdx.x;
  if (i < nwords) p[i] = 0u;
}

// x [B][T][168] f32 -> xh [T][B][192] f16 (zero padded)
__global__ void convert_x_kernel(const float* __restrict__ x, _Float16* __restrict__ xh) {
  size_t i = (size_t)blockIdx.x * 256 + threadIdx.x;     // total = T*B*192
  int k = (int)(i % KX0);
  size_t tb = i / KX0;
  int b = (int)(tb % BATCH);
  int t = (int)(tb / BATCH);
  float v = (k < IN0) ? x[((size_t)b * T_STEPS + t) * IN0 + k] : 0.0f;
  xh[i] = (_Float16)v;
}

// W0 packed: [dir][n][ Wih(168) | pad(24) | Whh(512) ] f16
__global__ void pack_w0_kernel(const float* __restrict__ Wih, const float* __restrict__ Whh,
                               _Float16* __restrict__ Wp) {
  size_t i = (size_t)blockIdx.x * 256 + threadIdx.x;     // total = 2*2048*704
  int k = (int)(i % KTOT0);
  size_t dn = i / KTOT0;
  float v;
  if (k < IN0)       v = Wih[dn * IN0 + k];
  else if (k < KX0)  v = 0.0f;
  else               v = Whh[dn * HID + (k - KX0)];
  Wp[i] = (_Float16)v;
}

// W1 packed: [dir][n][ Wih(1024) | Whh(512) ] f16
__global__ void pack_w1_kernel(const float* __restrict__ Wih, const float* __restrict__ Whh,
                               _Float16* __restrict__ Wp) {
  size_t i = (size_t)blockIdx.x * 256 + threadIdx.x;     // total = 2*2048*1536
  int k = (int)(i % KTOT1);
  size_t dn = i / KTOT1;
  float v = (k < KX1) ? Wih[dn * KX1 + k] : Whh[dn * HID + (k - KX1)];
  Wp[i] = (_Float16)v;
}

// Wout [168][1024] f32 -> [176][1024] f16 (zero rows)
__global__ void pack_wout_kernel(const float* __restrict__ Wo, _Float16* __restrict__ Wp) {
  size_t i = (size_t)blockIdx.x * 256 + threadIdx.x;     // total = 176*1024
  int k = (int)(i % (2 * HID));
  int n = (int)(i / (2 * HID));
  float v = (n < PRE) ? Wo[(size_t)n * (2 * HID) + k] : 0.0f;
  Wp[i] = (_Float16)v;
}

// ---------------- recurrent BiLSTM layer ----------------
// grid = 32 WGs: dir = blockIdx.x>>4, 16 WGs per dir each owning 32 hidden units.
// block = 256 (8 waves). Wave = (batch-tile mt, unit-tile ut) x all 4 gates, so
// i/f/g/o fusion is wave-local; c state lives in registers across all 512 steps.
__global__ __launch_bounds__(256)
void lstm_layer_kernel(const _Float16* __restrict__ xin, int Kx, int Ktot,
                       const _Float16* __restrict__ Wall, const float* __restrict__ bias,
                       _Float16* __restrict__ hbuf, _Float16* __restrict__ hout,
                       unsigned* __restrict__ ctr)
{
  __shared__ __align__(16) _Float16 Alds[BATCH * 512];   // 64 KB A-staging chunk
  const int tid  = threadIdx.x;
  const int lane = tid & 31;
  const int wave = tid >> 5;
  const int dir  = blockIdx.x >> 4;
  const int wg   = blockIdx.x & 15;
  const int unit0 = wg * 32;
  const int mt = wave & 3;          // batch tile (16 rows)
  const int ut = wave >> 2;         // unit tile (16 units), 0..1

  const _Float16* W  = Wall + (size_t)dir * NGATE * Ktot;
  const float*    bv = bias + dir * NGATE;
  _Float16*       hb = hbuf + dir * BATCH * HID;
  const int dirOff = dir * HID;

  const int crow = ((lane >> 4) << 3);   // C/D: M offset 0 or 8
  const int ccol = lane & 15;            // C/D: N = lane&15
  const int kh   = ((lane >> 4) << 4);   // A frag K offset (0/16)
  const int ko   = ((lane >> 4) << 4);   // B frag K offset (0/16)
  const int am   = mt * 16 + (lane & 15);
  const int unit = unit0 + ut * 16 + ccol;

  // per-lane bias + fixed per-gate B row base pointers (hoists 64-bit addr math)
  float bload[4];
  const _Float16* brow[4];
#pragma unroll
  for (int g = 0; g < 4; ++g) {
    bload[g] = bv[g * HID + unit];
    brow[g]  = W + (size_t)(g * HID + unit) * Ktot + ko;
  }

  float cst[8];
#pragma unroll
  for (int r = 0; r < 8; ++r) cst[r] = 0.0f;

  const unsigned NWG = gridDim.x;
  for (int step = 0; step < T_STEPS; ++step) {
    const int t = dir ? (T_STEPS - 1 - step) : step;
    const _Float16* xrow = xin + (size_t)t * BATCH * Kx;

    // prefetch next timestep's input rows into cache behind this step's WMMAs
    if (step + 1 < T_STEPS) {
      const int tn = dir ? (T_STEPS - 2 - step) : step + 1;
      const _Float16* xnext = xin + (size_t)tn * BATCH * Kx;
      for (int i = tid; i < (BATCH * Kx) / 64; i += 256)
        __builtin_prefetch(xnext + (size_t)i * 64, 0, 1);
    }

    v8f acc[4];
#pragma unroll
    for (int g = 0; g < 4; ++g) {
      v8f a;
#pragma unroll
      for (int r = 0; r < 8; ++r) a[r] = bload[g];
      acc[g] = a;
    }

    int kdone = 0;
    while (kdone < Ktot) {
      int chunk = Ktot - kdone; if (chunk > 512) chunk = 512;
      int nblk = chunk >> 3;     // 16-byte blocks per row
      for (int i = tid; i < BATCH * nblk; i += 256) {
        int row = i / nblk, cb = i - row * nblk;
        int gc = kdone + cb * 8;
        const _Float16* src = (gc < Kx) ? (xrow + (size_t)row * Kx + gc)
                                        : (hb + row * HID + (gc - Kx));
        _Float16* dst = Alds + row * 512 + cb * 8;
#if HAVE_ASYNC_LDS
        __builtin_amdgcn_global_load_async_to_lds_b128(
            (gas_ptr)(uintptr_t)src, (las_ptr)(uintptr_t)dst, 0, 0);
#else
        *(v8h*)dst = *(const v8h*)src;
#endif
      }
#if HAVE_ASYNC_LDS
      __builtin_amdgcn_s_wait_asynccnt(0);
#endif
      __syncthreads();
#pragma unroll 2
      for (int kk = 0; kk < chunk; kk += 32) {
        const _Float16* ap = Alds + am * 512 + kk + kh;
        v16h afrag = join8(*(const v8h*)ap, *(const v8h*)(ap + 8));
#pragma unroll
        for (int g = 0; g < 4; ++g) {
          const _Float16* bp = brow[g] + kdone + kk;
          v16h bfrag = join8(*(const v8h*)bp, *(const v8h*)(bp + 8));
          acc[g] = __builtin_amdgcn_wmma_f32_16x16x32_f16(
              false, afrag, false, bfrag, (short)0, acc[g], false, false);
        }
      }
      __syncthreads();
      kdone += chunk;
    }

    // gate fusion + state update + h writeback (f16)
#pragma unroll
    for (int r = 0; r < 8; ++r) {
      float iv = sigm(acc[0][r]);
      float fv = sigm(acc[1][r]);
      float gv = tanh_fast(acc[2][r]);
      float ov = sigm(acc[3][r]);
      float c  = fv * cst[r] + iv * gv;
      cst[r] = c;
      float h  = ov * tanh_fast(c);
      int b = mt * 16 + r + crow;
      _Float16 hh = (_Float16)h;
      hb[b * HID + unit] = hh;
      hout[((size_t)t * BATCH + b) * (2 * HID) + dirOff + unit] = hh;
    }

    // device-scope step barrier (32 co-resident WGs)
    __threadfence();
    __syncthreads();
    if (tid == 0) {
      atomicAdd(ctr, 1u);
      unsigned target = NWG * (unsigned)(step + 1);
      while (__hip_atomic_load(ctr, __ATOMIC_ACQUIRE, __HIP_MEMORY_SCOPE_AGENT) < target)
        __builtin_amdgcn_s_sleep(1);
    }
    __syncthreads();
  }
}

// ---------------- output projection: out[B,T,168] = h2 @ Wout^T + bout ----------------
__global__ __launch_bounds__(256)
void out_gemm_kernel(const _Float16* __restrict__ h2, const _Float16* __restrict__ Wo,
                     const float* __restrict__ bout, float* __restrict__ out)
{
  const int lane = threadIdx.x & 31;
  const int wave = threadIdx.x >> 5;
  int tile = blockIdx.x * 8 + wave;
  const int NT = NPAD / 16;                 // 11 N-tiles
  int mtile = tile / NT;
  int ntile = tile - mtile * NT;
  if (mtile >= (T_STEPS * BATCH) / 16) return;

  const int kh = ((lane >> 4) << 4);
  const int ko = ((lane >> 4) << 4);
  const int p  = ntile * 16 + (lane & 15);
  float bvv = (p < PRE) ? bout[p] : 0.0f;
  v8f acc;
#pragma unroll
  for (int r = 0; r < 8; ++r) acc[r] = bvv;

  const _Float16* arow = h2 + ((size_t)mtile * 16 + (lane & 15)) * (2 * HID);
  const _Float16* brow = Wo + (size_t)p * (2 * HID);
#pragma unroll 4
  for (int k = 0; k < 2 * HID; k += 32) {
    v16h a = join8(*(const v8h*)(arow + k + kh), *(const v8h*)(arow + k + kh + 8));
    v16h b = join8(*(const v8h*)(brow + k + ko), *(const v8h*)(brow + k + ko + 8));
    acc = __builtin_amdgcn_wmma_f32_16x16x32_f16(false, a, false, b, (short)0, acc, false, false);
  }
  if (p < PRE) {
#pragma unroll
    for (int r = 0; r < 8; ++r) {
      int fm = mtile * 16 + r + ((lane >> 4) << 3);
      int tt = fm >> 6;                     // fm = t*64 + b
      int bb = fm & 63;
      out[((size_t)bb * T_STEPS + tt) * PRE + p] = acc[r];
    }
  }
}

extern "C" void kernel_launch(void* const* d_in, const int* in_sizes, int n_in,
                              void* d_out, int out_size, void* d_ws, size_t ws_size,
                              hipStream_t stream) {
  (void)in_sizes; (void)n_in; (void)out_size; (void)ws_size;
  const float* x    = (const float*)d_in[0];
  const float* Wih0 = (const float*)d_in[1];
  const float* Whh0 = (const float*)d_in[2];
  const float* b0   = (const float*)d_in[3];
  const float* Wih1 = (const float*)d_in[4];
  const float* Whh1 = (const float*)d_in[5];
  const float* b1   = (const float*)d_in[6];
  const float* Wout = (const float*)d_in[7];
  const float* bout = (const float*)d_in[8];
  float* out = (float*)d_out;

  char* ws = (char*)d_ws;
  _Float16* xh    = (_Float16*)(ws + OFF_XH);
  _Float16* w0    = (_Float16*)(ws + OFF_W0);
  _Float16* w1    = (_Float16*)(ws + OFF_W1);
  _Float16* wo    = (_Float16*)(ws + OFF_WOUT);
  _Float16* hbuf0 = (_Float16*)(ws + OFF_HBUF0);
  _Float16* hbuf1 = (_Float16*)(ws + OFF_HBUF1);
  unsigned* ctr   = (unsigned*)(ws + OFF_CTR);
  _Float16* h1    = (_Float16*)(ws + OFF_H1);
  _Float16* h2    = (_Float16*)(ws + OFF_H2);

  // zero hbuf0 + hbuf1 + counters (contiguous region)
  const unsigned zwords = (131072u * 2u + 256u) / 4u;   // 65600
  lstm_zero_kernel<<<(zwords + 255u) / 256u, 256, 0, stream>>>((unsigned*)(ws + OFF_HBUF0), zwords);

  convert_x_kernel<<<(unsigned)((size_t)T_STEPS * BATCH * KX0 / 256), 256, 0, stream>>>(x, xh);
  pack_w0_kernel  <<<(unsigned)((size_t)2 * NGATE * KTOT0 / 256), 256, 0, stream>>>(Wih0, Whh0, w0);
  pack_w1_kernel  <<<(unsigned)((size_t)2 * NGATE * KTOT1 / 256), 256, 0, stream>>>(Wih1, Whh1, w1);
  pack_wout_kernel<<<(unsigned)((size_t)NPAD * 2 * HID / 256), 256, 0, stream>>>(Wout, wo);

  lstm_layer_kernel<<<32, 256, 0, stream>>>(xh, KX0, KTOT0, w0, b0, hbuf0, h1, ctr);
  lstm_layer_kernel<<<32, 256, 0, stream>>>(h1, KX1, KTOT1, w1, b1, hbuf1, h2, ctr + 16);

  const unsigned ntiles = (T_STEPS * BATCH / 16) * (NPAD / 16);   // 22528
  out_gemm_kernel<<<ntiles / 8, 256, 0, stream>>>(h2, wo, bout, out);
}
